// MultiHeadSelfAttention_46359876993065
// MI455X (gfx1250) — matrix operand order repaired and checked by reading, not compiled, not run on hardware
//
#include <hip/hip_runtime.h>

// ---------------------------------------------------------------------------
// MI455X (gfx1250) fused multi-head self-attention, bf16 WMMA path.
//   k1: QKV projection  (bf16 wmma, f32 accum)  -> Q,K head-split, V transposed
//   k2: flash-attention (streaming softmax, async-to-LDS double buffering)
//   k3: output projection + bias -> fp32
// Workspace: 4 * B*H*M*64 bf16 = 64 MiB (Q, K, Vt, merged attn).
// ---------------------------------------------------------------------------

#define DEVINL __device__ __forceinline__

typedef __attribute__((ext_vector_type(16))) __bf16 v16bf;
typedef __attribute__((ext_vector_type(8)))  float  v8f;
typedef __attribute__((ext_vector_type(4)))  int    v4i;

constexpr int Bb = 8, Mm = 2048, DIN = 512, Hh = 8, Dd = 64, DH = 512;
constexpr int MT = Bb * Mm;              // 16384 total rows

union Frag { v16bf v; unsigned u[8]; unsigned short s[16]; };

DEVINL unsigned short f2bf(float f) {            // round-to-nearest-even
  union { float f; unsigned u; } x{f};
  unsigned r = x.u + 0x7FFFu + ((x.u >> 16) & 1u);
  return (unsigned short)(r >> 16);
}
// pack two f32 -> packed bf16x2 in one VALU op when hardware path exists
DEVINL unsigned pack2(float lo, float hi) {
#if __has_builtin(__builtin_amdgcn_cvt_pk_bf16_f32)
  auto p = __builtin_amdgcn_cvt_pk_bf16_f32(lo, hi);
  unsigned u; __builtin_memcpy(&u, &p, 4);
  return u;
#else
  unsigned u;
  asm("v_cvt_pk_bf16_f32 %0, %1, %2" : "=v"(u) : "v"(lo), "v"(hi));
  return u;
#endif
}
// store a packed bf16x2 to two (possibly non-adjacent) 16-bit slots
DEVINL void store2_b16(unsigned short* p0, unsigned short* p1, unsigned u) {
  p0[0] = (unsigned short)u;
  p1[0] = (unsigned short)(u >> 16);     // -> *_store_b16_d16_hi
}
DEVINL v8f vzero() { v8f z = {0.f,0.f,0.f,0.f,0.f,0.f,0.f,0.f}; return z; }

DEVINL v8f wmma_bf16(v16bf a, v16bf b, v8f c) {
  return __builtin_amdgcn_wmma_f32_16x16x32_bf16(false, a, false, b,
                                                 (short)0, c, false, false);
}

// ---- async global->LDS copy (16B), ASYNCcnt-tracked --------------------
DEVINL void async_cp16(unsigned short* lds_dst, const unsigned short* gsrc) {
#if __has_builtin(__builtin_amdgcn_global_load_async_to_lds_b128)
  __builtin_amdgcn_global_load_async_to_lds_b128(
      (__attribute__((address_space(1))) v4i*)(size_t)gsrc,
      (__attribute__((address_space(3))) v4i*)(unsigned)(size_t)lds_dst,
      0, 0);
#else
  asm volatile("global_load_async_to_lds_b128 %0, %1, off"
               :: "v"((unsigned)(size_t)lds_dst),
                  "v"((unsigned long long)(size_t)gsrc)
               : "memory");
#endif
}
DEVINL void async_wait0() {
#if __has_builtin(__builtin_amdgcn_s_wait_asynccnt)
  __builtin_amdgcn_s_wait_asynccnt(0);
#else
  asm volatile("s_wait_asynccnt 0" ::: "memory");
#endif
}

// A fragment (16x32, bf16) from row-major bf16 tile, pitch in elements.
DEVINL v16bf load_a_frag(const unsigned short* p0, int pitch, int row0, int k0) {
  int lane = threadIdx.x & 31;
  const unsigned short* p =
      p0 + (size_t)(row0 + (lane & 15)) * pitch + k0 + ((lane >> 4) * 8);
  Frag f;
#pragma unroll
  for (int i = 0; i < 4; ++i) f.u[i]     = *(const unsigned*)(p + 2 * i);
#pragma unroll
  for (int i = 0; i < 4; ++i) f.u[4 + i] = *(const unsigned*)(p + 16 + 2 * i);
  return f.v;
}

// B fragment (32x16, bf16) from an n-major tile Bt[n][k], pitch in elements.
DEVINL v16bf load_bt_frag(const unsigned short* p0, int pitch, int n0, int k0) {
  int lane = threadIdx.x & 31;
  const unsigned short* p =
      p0 + (size_t)(n0 + (lane & 15)) * pitch + k0 + ((lane >> 4) * 8);
  Frag f;
#pragma unroll
  for (int i = 0; i < 4; ++i) f.u[i]     = *(const unsigned*)(p + 2 * i);
#pragma unroll
  for (int i = 0; i < 4; ++i) f.u[4 + i] = *(const unsigned*)(p + 16 + 2 * i);
  return f.v;
}

// ------------------------------ kernel 1 -----------------------------------
// Block tile 256 rows x 64 cols; wave tile 32x64 (8 WMMA / K-step).
__global__ __launch_bounds__(256)
void qkv_kernel(const float* __restrict__ x,  const float* __restrict__ Wq,
                const float* __restrict__ Wk, const float* __restrict__ Wv,
                unsigned short* __restrict__ Qo,   // [B,H,M,64]
                unsigned short* __restrict__ Ko,   // [B,H,M,64]
                unsigned short* __restrict__ Vo)   // [B,H,64,M] (transposed)
{
  constexpr int AP = 40, BP = 40;
  __shared__ unsigned short Als[256 * AP];         // 256 rows x 32 k
  __shared__ unsigned short Bls[64 * BP];          // 64 n x 32 k

  const int tid = threadIdx.x, wv = tid >> 5, lane = tid & 31;
  const int region = blockIdx.y >> 3;              // 0=Q 1=K 2=V
  const int h  = blockIdx.y & 7;
  const int row0 = blockIdx.x * 256;
  const int c0 = h * 64;
  const float* W = (region == 0) ? Wq : (region == 1) ? Wk : Wv;

  v8f acc[2][4] = {{vzero(),vzero(),vzero(),vzero()},
                   {vzero(),vzero(),vzero(),vzero()}};

  for (int ks = 0; ks < DIN / 32; ++ks) {
    const int k0 = ks * 32;
    __syncthreads();
#pragma unroll
    for (int p = 0; p < 2; ++p) {  // X tile fp32->bf16, 256x32
      int idx = (tid + p * 256) * 16, r = idx >> 5, c = idx & 31;
      const float* src = x + (size_t)(row0 + r) * DIN + k0 + c;
      unsigned short* dst = Als + r * AP + c;
#pragma unroll
      for (int i = 0; i < 8; ++i)
        *(unsigned*)(dst + 2 * i) = pack2(src[2 * i], src[2 * i + 1]);
    }
    { // W tile as Bt[n][k], packed conversion, split stores
      int k = tid >> 3, nb = (tid & 7) * 8;
      const float* src = W + (size_t)(k0 + k) * DH + c0 + nb;
#pragma unroll
      for (int e = 0; e < 8; e += 2) {
        unsigned u = pack2(src[e], src[e + 1]);
        store2_b16(&Bls[(nb + e) * BP + k], &Bls[(nb + e + 1) * BP + k], u);
      }
    }
    __syncthreads();

    v16bf a0 = load_a_frag(Als, AP, wv * 32, 0);
    v16bf a1 = load_a_frag(Als, AP, wv * 32 + 16, 0);
#pragma unroll
    for (int nt = 0; nt < 4; ++nt) {
      v16bf b = load_bt_frag(Bls, BP, nt * 16, 0);
      acc[0][nt] = wmma_bf16(a0, b, acc[0][nt]);
      acc[1][nt] = wmma_bf16(a1, b, acc[1][nt]);
    }
  }

  const int hi = lane >> 4, n = lane & 15;
#pragma unroll
  for (int t = 0; t < 2; ++t) {
    const int mrow = row0 + wv * 32 + t * 16 + hi * 8;
    const int b = mrow / Mm, ml = mrow % Mm;
    if (region < 2) {
      unsigned short* dst = (region == 0) ? Qo : Ko;
#pragma unroll
      for (int nt = 0; nt < 4; ++nt) {
        int d = nt * 16 + n;
        size_t base = ((size_t)(b * Hh + h) * Mm + ml) * Dd + d;
#pragma unroll
        for (int r = 0; r < 8; r += 2) {
          unsigned u = pack2(acc[t][nt][r], acc[t][nt][r + 1]);
          store2_b16(&dst[base + (size_t)r * Dd],
                     &dst[base + (size_t)(r + 1) * Dd], u);
        }
      }
    } else {
#pragma unroll
      for (int nt = 0; nt < 4; ++nt) {
        int d = nt * 16 + n;
        unsigned* dp = (unsigned*)(Vo + ((size_t)(b * Hh + h) * Dd + d) * Mm + ml);
#pragma unroll
        for (int r = 0; r < 4; ++r)
          dp[r] = pack2(acc[t][nt][2 * r], acc[t][nt][2 * r + 1]);
      }
    }
  }
}

// ------------------------------ kernel 2 -----------------------------------
__global__ __launch_bounds__(256)
void attn_kernel(const unsigned short* __restrict__ Qb,
                 const unsigned short* __restrict__ Kb,
                 const unsigned short* __restrict__ Vt,
                 unsigned short* __restrict__ Ao)   // merged [B,M,512]
{
  constexpr int KP = 80, VP = 40, PP = 40;
  __shared__ unsigned short Kt[2][32 * KP];        // double-buffered K [j][d]
  __shared__ unsigned short Vs[2][64 * VP];        // double-buffered V [d][j]
  __shared__ unsigned short Ps[8][16 * PP];        // per-wave P transpose pad

  const int tid = threadIdx.x, wv = tid >> 5, lane = tid & 31;
  const int hi = lane >> 4, n = lane & 15;
  const int bh = blockIdx.y;
  const int q0 = blockIdx.x * 128 + wv * 16;

  const unsigned short* Qh = Qb + (size_t)bh * Mm * Dd;
  const unsigned short* Kh = Kb + (size_t)bh * Mm * Dd;
  const unsigned short* Vh = Vt + (size_t)bh * Dd * Mm;

  // async stage of one 32-key K/V tile into LDS buffer `buf`
  auto stage = [&](int buf, int j0) {
    int j = tid >> 3, cbk = (tid & 7) * 8;
    async_cp16(&Kt[buf][j * KP + cbk], Kh + (size_t)(j0 + j) * Dd + cbk);
    int d = tid >> 2, cbv = (tid & 3) * 8;
    async_cp16(&Vs[buf][d * VP + cbv], Vh + (size_t)d * Mm + j0 + cbv);
  };

  v16bf qlo = load_a_frag(Qh, Dd, q0, 0);
  v16bf qhi = load_a_frag(Qh, Dd, q0, 32);

  const float c = 22.627416997969522f;             // 1/SCALE = sqrt(512) = sqrt(DH)
  float mrun[8], lrun[8];
#pragma unroll
  for (int r = 0; r < 8; ++r) { mrun[r] = -3.0e38f; lrun[r] = 0.0f; }
  v8f o[4] = {vzero(), vzero(), vzero(), vzero()};

  stage(0, 0);                                     // prologue prefetch
  for (int kt = 0; kt < Mm / 32; ++kt) {
    const int cur = kt & 1;
    async_wait0();                                 // my tile-`cur` loads done
    __syncthreads();                               // everyone's done; prev reads done
    if (kt + 1 < Mm / 32) stage(cur ^ 1, (kt + 1) * 32);   // overlap next tile

    const unsigned short* Kc = &Kt[cur][0];
    const unsigned short* Vc = &Vs[cur][0];

    // S = Q K^T for 16 queries x 32 keys (K-dim 64 = 2 WMMAs per half)
    v8f s0 = vzero(), s1 = vzero();
    s0 = wmma_bf16(qlo, load_bt_frag(Kc, KP, 0,  0),  s0);
    s0 = wmma_bf16(qhi, load_bt_frag(Kc, KP, 0,  32), s0);
    s1 = wmma_bf16(qlo, load_bt_frag(Kc, KP, 16, 0),  s1);
    s1 = wmma_bf16(qhi, load_bt_frag(Kc, KP, 16, 32), s1);

    // online softmax (C layout: rows 0-7 in lanes 0-15, 8-15 in lanes 16-31)
    float alpha[8];
#pragma unroll
    for (int r = 0; r < 8; ++r) {
      float v0 = s0[r] * c, v1 = s1[r] * c;
      float mx = fmaxf(v0, v1);
      mx = fmaxf(mx, __shfl_xor(mx, 1, 32));
      mx = fmaxf(mx, __shfl_xor(mx, 2, 32));
      mx = fmaxf(mx, __shfl_xor(mx, 4, 32));
      mx = fmaxf(mx, __shfl_xor(mx, 8, 32));
      float mnew = fmaxf(mrun[r], mx);
      alpha[r] = __expf(mrun[r] - mnew);
      float p0 = __expf(v0 - mnew), p1 = __expf(v1 - mnew);
      s0[r] = p0; s1[r] = p1;
      float rs = p0 + p1;
      rs += __shfl_xor(rs, 1, 32);
      rs += __shfl_xor(rs, 2, 32);
      rs += __shfl_xor(rs, 4, 32);
      rs += __shfl_xor(rs, 8, 32);
      lrun[r] = lrun[r] * alpha[r] + rs;
      mrun[r] = mnew;
    }
#pragma unroll
    for (int dt = 0; dt < 4; ++dt)
#pragma unroll
      for (int r = 0; r < 8; ++r) o[dt][r] *= alpha[r];

    // transpose P (C layout -> A layout) via wave-private LDS scratch;
    // pack the two tile values per row into one cvt, split b16 stores
    unsigned short* ps = &Ps[wv][0];
#pragma unroll
    for (int r = 0; r < 8; ++r) {
      unsigned u = pack2(s0[r], s1[r]);
      store2_b16(&ps[(hi * 8 + r) * PP + n], &ps[(hi * 8 + r) * PP + 16 + n], u);
    }
    asm volatile("s_wait_dscnt 0" ::: "memory");   // cross-lane LDS RAW
    v16bf pa = load_a_frag(ps, PP, 0, 0);
#pragma unroll
    for (int dt = 0; dt < 4; ++dt)
      o[dt] = wmma_bf16(pa, load_bt_frag(Vc, VP, dt * 16, 0), o[dt]);
  }

  // normalize and write merged layout [b, m, h*64+d]
  const int b = bh >> 3, h = bh & 7;
#pragma unroll
  for (int dt = 0; dt < 4; ++dt) {
#pragma unroll
    for (int r = 0; r < 8; r += 2) {
      int m = q0 + hi * 8 + r;
      unsigned u = pack2(o[dt][r] / lrun[r], o[dt][r + 1] / lrun[r + 1]);
      size_t base = ((size_t)b * Mm + m) * DH + h * 64 + dt * 16 + n;
      store2_b16(&Ao[base], &Ao[base + DH], u);
    }
  }
}

// ------------------------------ kernel 3 -----------------------------------
__global__ __launch_bounds__(256)
void out_kernel(const unsigned short* __restrict__ Am,
                const float* __restrict__ Wo, const float* __restrict__ bo,
                float* __restrict__ out)
{
  constexpr int AP = 40, BP = 40;
  __shared__ unsigned short Als[256 * AP];
  __shared__ unsigned short Bls[64 * BP];
  const int tid = threadIdx.x, wv = tid >> 5, lane = tid & 31;
  const int row0 = blockIdx.x * 256;
  const int c0 = blockIdx.y * 64;

  v8f acc[2][4] = {{vzero(),vzero(),vzero(),vzero()},
                   {vzero(),vzero(),vzero(),vzero()}};
  for (int ks = 0; ks < DH / 32; ++ks) {
    const int k0 = ks * 32;
    __syncthreads();
#pragma unroll
    for (int p = 0; p < 2; ++p) {  // A tile already bf16 -> async straight copy
      int idx = (tid + p * 256) * 16, r = idx >> 5, cc = idx & 31;
      async_cp16(Als + r * AP + cc, Am + (size_t)(row0 + r) * DH + k0 + cc);
      async_cp16(Als + r * AP + cc + 8,
                 Am + (size_t)(row0 + r) * DH + k0 + cc + 8);
    }
    { // Wo tile as Bt[n][k]
      int k = tid >> 3, nb = (tid & 7) * 8;
      const float* src = Wo + (size_t)(k0 + k) * DIN + c0 + nb;
#pragma unroll
      for (int e = 0; e < 8; e += 2) {
        unsigned u = pack2(src[e], src[e + 1]);
        store2_b16(&Bls[(nb + e) * BP + k], &Bls[(nb + e + 1) * BP + k], u);
      }
    }
    async_wait0();
    __syncthreads();
    v16bf a0 = load_a_frag(Als, AP, wv * 32, 0);
    v16bf a1 = load_a_frag(Als, AP, wv * 32 + 16, 0);
#pragma unroll
    for (int nt = 0; nt < 4; ++nt) {
      v16bf b = load_bt_frag(Bls, BP, nt * 16, 0);
      acc[0][nt] = wmma_bf16(a0, b, acc[0][nt]);
      acc[1][nt] = wmma_bf16(a1, b, acc[1][nt]);
    }
  }

  const int hi = lane >> 4, n = lane & 15;
#pragma unroll
  for (int t = 0; t < 2; ++t) {
    const int mrow = row0 + wv * 32 + t * 16 + hi * 8;
#pragma unroll
    for (int nt = 0; nt < 4; ++nt) {
      int cg = c0 + nt * 16 + n;
      float bias = bo[cg];
#pragma unroll
      for (int r = 0; r < 8; ++r)
        out[(size_t)(mrow + r) * DIN + cg] = acc[t][nt][r] + bias;
    }
  }
}

// ------------------------------ launcher -----------------------------------
extern "C" void kernel_launch(void* const* d_in, const int* in_sizes, int n_in,
                              void* d_out, int out_size, void* d_ws, size_t ws_size,
                              hipStream_t stream)
{
  (void)in_sizes; (void)n_in; (void)out_size; (void)ws_size;
  const float* x  = (const float*)d_in[0];
  const float* Wq = (const float*)d_in[1];
  const float* Wk = (const float*)d_in[2];
  const float* Wv = (const float*)d_in[3];
  const float* Wo = (const float*)d_in[4];
  const float* bo = (const float*)d_in[5];
  float* out = (float*)d_out;

  const size_t nEl = (size_t)Bb * Hh * Mm * Dd;    // 8,388,608 bf16 each
  unsigned short* Qw = (unsigned short*)d_ws;
  unsigned short* Kw = Qw + nEl;
  unsigned short* Vw = Kw + nEl;
  unsigned short* Aw = Vw + nEl;                   // total 64 MiB workspace

  qkv_kernel <<<dim3(MT / 256, 24),        256, 0, stream>>>(x, Wq, Wk, Wv, Qw, Kw, Vw);
  attn_kernel<<<dim3(Mm / 128, Bb * Hh),   256, 0, stream>>>(Qw, Kw, Vw, Aw);
  out_kernel <<<dim3(MT / 256, DIN / 64),  256, 0, stream>>>(Aw, Wo, bo, out);
}
